// GCNLayer_72696616452752
// MI455X (gfx1250) — compile-verified, matrix operand order checked by reading, not run on hardware
//
#include <hip/hip_runtime.h>
#include <hip/hip_bf16.h>

typedef __attribute__((ext_vector_type(16))) _Float16 v16h;
typedef __attribute__((ext_vector_type(8)))  float    v8f;

#define N_NODES_C 100000
#define N_RELS_C  500
#define N_EDGES_C 1000000
#define IN_DIM_C  64
#define OUT_DIM_C 64
#define K_DIM_C   128
#define BN_EPS_C  1e-5f

// workspace byte offsets (all 256B aligned)
#define OFF_BFRAG 0u          // 16 tiles * 32 lanes * 32B = 16 KB (reserve 32 KB)
#define OFF_ATTEN 32768u      // 1M floats   (reserve 4 MB)
#define OFF_COEFF 4227072u    // 100k floats (reserve 512 KB)
#define OFF_XACC  4751360u    // 6.4M floats (25.6 MB)
#define OFF_STATS 30351360u   // 128 floats  (reserve 1 KB)
#define OFF_RTMP  30352384u   // 32k floats  (128 KB)

// ---------------------------------------------------------------------------
// Pack W_m^T (B matrix, K=128 x N=64) into CDNA5 f16 B fragments.
// Fragment index [t*4+s][lane]; lane holds column n = 16t + (lane&15),
// halves 0..7 -> K = 32s + 8*(lane>>4) + j, halves 8..15 -> +16.
// ---------------------------------------------------------------------------
__global__ void prep_b_kernel(const float* __restrict__ Wm, v16h* __restrict__ bfrag) {
    int lane  = threadIdx.x;        // launched with 32 threads
    int group = lane >> 4;
    int m     = lane & 15;
    for (int t = 0; t < 4; ++t) {
        int n = 16 * t + m;
        for (int s = 0; s < 4; ++s) {
            int kb = 32 * s + 8 * group;
            v16h b;
#pragma unroll
            for (int j = 0; j < 8; ++j) {
                // B[k][n] = W_m[n][k], W_m row-major (64 x 128)
                b[j]     = (_Float16)Wm[n * K_DIM_C + kb + j];
                b[j + 8] = (_Float16)Wm[n * K_DIM_C + kb + 16 + j];
            }
            bfrag[(t * 4 + s) * 32 + lane] = b;
        }
    }
}

__global__ void zerof_kernel(float* __restrict__ p, int n) {
    int i = blockIdx.x * blockDim.x + threadIdx.x;
    if (i < n) p[i] = 0.f;
}

// ---------------------------------------------------------------------------
// One wave computes the 16-edge x 64-out message tile (K=128) with WMMA and
// writes it (bias added) to this wave's LDS tile (stride 65 to dodge banks).
// ---------------------------------------------------------------------------
__device__ __forceinline__ void compute_msg_tile(
    int e0, int lane,
    const float* __restrict__ x, const float* __restrict__ r,
    const int* __restrict__ srcIdx, const int* __restrict__ relIdx,
    const v16h* __restrict__ bfrag, const float* __restrict__ bm,
    float* __restrict__ tile)
{
    int group = lane >> 4;
    int m     = lane & 15;
    int e     = e0 + m;                         // tile fully in-range (1M % 16 == 0)
    const float* srcRow = x + (size_t)srcIdx[e] * IN_DIM_C;
    const float* relRow = r + (size_t)relIdx[e] * IN_DIM_C;

    // A fragments: concat feature row (x | r), documented 16-bit A layout.
    v16h afrag[4];
#pragma unroll
    for (int s = 0; s < 4; ++s) {
        int kb = 32 * s + 8 * group;
#pragma unroll
        for (int j = 0; j < 8; ++j) {
            int k0 = kb + j;
            int k1 = kb + 16 + j;
            float v0 = (k0 < IN_DIM_C) ? srcRow[k0] : relRow[k0 - IN_DIM_C];
            float v1 = (k1 < IN_DIM_C) ? srcRow[k1] : relRow[k1 - IN_DIM_C];
            afrag[s][j]     = (_Float16)v0;
            afrag[s][j + 8] = (_Float16)v1;
        }
    }

#pragma unroll
    for (int t = 0; t < 4; ++t) {
        v8f c = {0.f, 0.f, 0.f, 0.f, 0.f, 0.f, 0.f, 0.f};
#pragma unroll
        for (int s = 0; s < 4; ++s) {
            v16h b = bfrag[(t * 4 + s) * 32 + lane];
            c = __builtin_amdgcn_wmma_f32_16x16x32_f16(
                    false, afrag[s], false, b, (short)0, c, false, false);
        }
        int   n    = 16 * t + m;
        float bias = bm[n];
#pragma unroll
        for (int i = 0; i < 8; ++i) {
            int row = i + 8 * group;            // C layout: vgpr i, lane half -> M
            tile[row * 65 + n] = c[i] + bias;
        }
    }
    // LDS tile is read cross-lane next: drain DS ops for this wave.
    asm volatile("s_wait_dscnt 0x0" ::: "memory");
}

// ---------------------------------------------------------------------------
// Pass 1: attention logits -> exp(tanh) -> atomic segment sum of coefficients.
// ---------------------------------------------------------------------------
__global__ void __launch_bounds__(256) pass1_kernel(
    const float* __restrict__ x, const float* __restrict__ r,
    const int* __restrict__ src, const int* __restrict__ rel,
    const int* __restrict__ tgt,
    const float* __restrict__ attw, const float* __restrict__ que,
    const v16h* __restrict__ bfrag, const float* __restrict__ bm,
    float* __restrict__ atten, float* __restrict__ coeffs, int nEdges)
{
    __shared__ float tiles[8][16 * 65];
    int wave = threadIdx.x >> 5;
    int lane = threadIdx.x & 31;
    int e0   = ((blockIdx.x << 3) + wave) << 4;     // 16 edges per wave
    if (e0 >= nEdges) return;                       // wave-uniform

    float* tile = tiles[wave];
    compute_msg_tile(e0, lane, x, r, src, rel, bfrag, bm, tile);

    float qdot = 0.f;
#pragma unroll 8
    for (int k = 0; k < OUT_DIM_C; ++k) qdot += attw[OUT_DIM_C + k] * que[k];

    if (lane < 16) {
        int e = e0 + lane;
        const float* row = tile + lane * 65;
        float d = qdot;
#pragma unroll 8
        for (int k = 0; k < OUT_DIM_C; ++k) d += row[k] * attw[k];
        float a = __expf(tanhf(d));
        atten[e] = a;
        atomicAdd(&coeffs[tgt[e]], a);
    }
}

// ---------------------------------------------------------------------------
// Pass 2: recompute messages, weight by atten/coeff[tgt], scatter-add.
// ---------------------------------------------------------------------------
__global__ void __launch_bounds__(256) pass2_kernel(
    const float* __restrict__ x, const float* __restrict__ r,
    const int* __restrict__ src, const int* __restrict__ rel,
    const int* __restrict__ tgt,
    const v16h* __restrict__ bfrag, const float* __restrict__ bm,
    const float* __restrict__ atten, const float* __restrict__ coeffs,
    float* __restrict__ xacc, int nEdges)
{
    __shared__ float tiles[8][16 * 65];
    int wave = threadIdx.x >> 5;
    int lane = threadIdx.x & 31;
    int e0   = ((blockIdx.x << 3) + wave) << 4;
    if (e0 >= nEdges) return;                       // wave-uniform

    float* tile = tiles[wave];
    compute_msg_tile(e0, lane, x, r, src, rel, bfrag, bm, tile);

#pragma unroll 4
    for (int i = 0; i < 16; ++i) {
        int   e = e0 + i;
        int   t = tgt[e];
        float w = atten[e] / coeffs[t];
        float v0 = tile[i * 65 + lane];
        float v1 = tile[i * 65 + lane + 32];
        atomicAdd(&xacc[(size_t)t * OUT_DIM_C + lane],      v0 * w);
        atomicAdd(&xacc[(size_t)t * OUT_DIM_C + lane + 32], v1 * w);
    }
}

// ---------------------------------------------------------------------------
// Column stats (sum / sumsq per output dim) with LDS block reduction.
// ---------------------------------------------------------------------------
__global__ void __launch_bounds__(256) colstats_kernel(
    const float* __restrict__ xacc, float* __restrict__ stats, int nNodes)
{
    __shared__ float bs[128];
    int tid = threadIdx.x;
    if (tid < 128) bs[tid] = 0.f;
    __syncthreads();

    int col       = tid & 63;
    int rowStart  = (blockIdx.x * blockDim.x + tid) >> 6;
    int rowStride = (gridDim.x * blockDim.x) >> 6;
    float s = 0.f, sq = 0.f;
    for (int row = rowStart; row < nNodes; row += rowStride) {
        float v = xacc[(size_t)row * OUT_DIM_C + col];
        s += v; sq += v * v;
    }
    atomicAdd(&bs[col], s);
    atomicAdd(&bs[64 + col], sq);
    __syncthreads();
    if (tid < 128) atomicAdd(&stats[tid], bs[tid]);
}

__global__ void __launch_bounds__(256) finalize_x_kernel(
    const float* __restrict__ xacc, const float* __restrict__ stats,
    const float* __restrict__ gamma, const float* __restrict__ beta,
    float* __restrict__ out, int n)
{
    int i = blockIdx.x * blockDim.x + threadIdx.x;
    if (i >= n) return;
    int   col = i & 63;
    float m   = stats[col] * (1.f / (float)N_NODES_C);
    float var = stats[64 + col] * (1.f / (float)N_NODES_C) - m * m;
    var = fmaxf(var, 0.f);
    float v = (xacc[i] - m) * rsqrtf(var + BN_EPS_C) * gamma[col] + beta[col];
    out[i] = tanhf(v);
}

// ---------------------------------------------------------------------------
// r_new = tanh(BN(r @ W_r^T + b_r)); tiny: one block, one column per thread.
// ---------------------------------------------------------------------------
__global__ void rnew_kernel(
    const float* __restrict__ r, const float* __restrict__ Wr,
    const float* __restrict__ br,
    const float* __restrict__ gamma, const float* __restrict__ beta,
    float* __restrict__ rtmp, float* __restrict__ out)
{
    int o = threadIdx.x;                        // 64 threads
    float s = 0.f, sq = 0.f;
    const float* w = Wr + o * IN_DIM_C;
    for (int i = 0; i < N_RELS_C; ++i) {
        const float* ri = r + i * IN_DIM_C;
        float d = br[o];
#pragma unroll 8
        for (int k = 0; k < IN_DIM_C; ++k) d += ri[k] * w[k];
        rtmp[i * OUT_DIM_C + o] = d;
        s += d; sq += d * d;
    }
    float m   = s * (1.f / (float)N_RELS_C);
    float var = fmaxf(sq * (1.f / (float)N_RELS_C) - m * m, 0.f);
    float inv = rsqrtf(var + BN_EPS_C);
    float g = gamma[o], b = beta[o];
    for (int i = 0; i < N_RELS_C; ++i)
        out[i * OUT_DIM_C + o] = tanhf((rtmp[i * OUT_DIM_C + o] - m) * inv * g + b);
}

extern "C" void kernel_launch(void* const* d_in, const int* in_sizes, int n_in,
                              void* d_out, int out_size, void* d_ws, size_t ws_size,
                              hipStream_t stream) {
    const float* x    = (const float*)d_in[0];
    const float* r    = (const float*)d_in[1];
    const float* que  = (const float*)d_in[2];
    const int*   eidx = (const int*)d_in[3];
    const int*   eat  = (const int*)d_in[4];
    // d_in[5] = edge_type (unused by reference)
    const float* Wm   = (const float*)d_in[6];
    const float* bm   = (const float*)d_in[7];
    const float* attw = (const float*)d_in[8];
    const float* Wr   = (const float*)d_in[9];
    const float* br   = (const float*)d_in[10];
    const float* eg   = (const float*)d_in[11];
    const float* eb   = (const float*)d_in[12];
    const float* rg   = (const float*)d_in[13];
    const float* rb   = (const float*)d_in[14];

    const int nEdges = in_sizes[4];
    const int* src = eidx;
    const int* tgt = eidx + nEdges;

    char* ws = (char*)d_ws;
    v16h*  bfrag  = (v16h*)(ws + OFF_BFRAG);
    float* atten  = (float*)(ws + OFF_ATTEN);
    float* coeffs = (float*)(ws + OFF_COEFF);
    float* xacc   = (float*)(ws + OFF_XACC);
    float* stats  = (float*)(ws + OFF_STATS);
    float* rtmp   = (float*)(ws + OFF_RTMP);

    float* out_x = (float*)d_out;
    float* out_r = out_x + (size_t)N_NODES_C * OUT_DIM_C;

    // 0) prep constant B fragments + zero accumulators
    prep_b_kernel<<<1, 32, 0, stream>>>(Wm, bfrag);
    zerof_kernel<<<(N_NODES_C + 255) / 256, 256, 0, stream>>>(coeffs, N_NODES_C);
    zerof_kernel<<<(N_NODES_C * OUT_DIM_C + 255) / 256, 256, 0, stream>>>(
        xacc, N_NODES_C * OUT_DIM_C);
    zerof_kernel<<<1, 128, 0, stream>>>(stats, 128);

    // 1) attention pass  2) weighted scatter pass
    int nWaves  = (nEdges + 15) / 16;
    int nBlocks = (nWaves + 7) / 8;
    pass1_kernel<<<nBlocks, 256, 0, stream>>>(
        x, r, src, eat, tgt, attw, que, bfrag, bm, atten, coeffs, nEdges);
    pass2_kernel<<<nBlocks, 256, 0, stream>>>(
        x, r, src, eat, tgt, bfrag, bm, atten, coeffs, xacc, nEdges);

    // 3) batch-norm stats + finalize x_new
    colstats_kernel<<<128, 256, 0, stream>>>(xacc, stats, N_NODES_C);
    finalize_x_kernel<<<(N_NODES_C * OUT_DIM_C + 255) / 256, 256, 0, stream>>>(
        xacc, stats, eg, eb, out_x, N_NODES_C * OUT_DIM_C);

    // 4) r_new
    rnew_kernel<<<1, 64, 0, stream>>>(r, Wr, br, rg, rb, rtmp, out_r);
    (void)n_in; (void)out_size; (void)ws_size;
}
